// FusedMoEMethod_77129022701987
// MI455X (gfx1250) — matrix operand order repaired and checked by reading, not compiled, not run on hardware
//
#include <hip/hip_runtime.h>

// Problem constants (match reference): tokens, hidden, experts, intermediate
#define TOK  4096
#define HID  2048
#define NEXP 8
#define IMID 5632

typedef __bf16 bf16;
typedef __attribute__((ext_vector_type(16))) __bf16 v16bf;
typedef __attribute__((ext_vector_type(8)))  __bf16 v8bf;
typedef __attribute__((ext_vector_type(8)))  float  v8f;
typedef __attribute__((ext_vector_type(4)))  float  v4f;

// LDS row stride in bf16 elements for a 32-wide K tile; 40*2B = 80B stride
// spreads the 16B fragment reads of lanes 0..15 across all 64 banks.
#define LDT 40

// ---------------------------------------------------------------------------
// CDNA5-specific primitives
// ---------------------------------------------------------------------------
__device__ inline v8f mma_bf16(v16bf a, v16bf b, v8f c) {
  // (neg_a, A, neg_b, B, c_mod, C, reuse_a, reuse_b)
  return __builtin_amdgcn_wmma_f32_16x16x32_bf16(false, a, false, b,
                                                 (short)0, c, false, false);
}

// Async copy 16B global -> LDS per lane (GLOBAL_LOAD_ASYNC_TO_LDS_B128,
// tracked by ASYNCcnt; ISA 08_async_tensor §4). lds = LDS byte offset
// (low 32 bits of the generic shared address), g = per-lane global address.
__device__ inline void async_ld16(unsigned lds, const bf16* g) {
  asm volatile("global_load_async_to_lds_b128 %0, %1, off"
               :: "v"(lds), "v"(g)
               : "memory");
}
__device__ inline void wait_async0() {
  asm volatile("s_wait_asynccnt 0x0" ::: "memory");
}
__device__ inline unsigned lds_off(const void* p) {
  // Generic pointers into LDS carry the offset in the low 32 bits (aperture
  // lives in addr[63:32]; see ISA 10.2).
  return (unsigned)(size_t)p;
}

// A-matrix 16x32 bf16 fragment (ISA 7.12.2): lane<16 holds row (lane&15),
// K = 0..7 then 16..23; lane>=16 holds K = 8..15 then 24..31.
__device__ inline v16bf frag_a(const bf16* tile, int row0, int lane) {
  const bf16* p = tile + (row0 + (lane & 15)) * LDT;
  const int hi = (lane >> 4) << 3;  // 0 or 8
  union { v16bf v; v8bf h[2]; } u;
  u.h[0] = *(const v8bf*)(p + hi);
  u.h[1] = *(const v8bf*)(p + 16 + hi);
  return u.v;
}

// B-matrix 32x16 bf16 fragment: lane (l&15) holds column n = l&15 of B,
// i.e. row n of the row-major weight tile; lanes 0..15 hold K=0..7 then
// 16..23, lanes 16..31 hold K=8..15 then 24..31 (same as A; two 16B LDS reads).
__device__ inline v16bf frag_b(const bf16* tile, int row0, int lane) {
  const bf16* p = tile + (row0 + (lane & 15)) * LDT + ((lane >> 4) << 3);
  union { v16bf v; v8bf h[2]; } u;
  u.h[0] = *(const v8bf*)(p);
  u.h[1] = *(const v8bf*)(p + 16);
  return u.v;
}

// ---------------------------------------------------------------------------
// 0) Bulk fp32 -> bf16 conversion (one-time, bandwidth-bound)
// ---------------------------------------------------------------------------
__global__ void cvt_kernel(const float* __restrict__ src,
                           bf16* __restrict__ dst, size_t n) {
  size_t i = ((size_t)blockIdx.x * blockDim.x + threadIdx.x) * 8;
  const size_t stride = (size_t)gridDim.x * blockDim.x * 8;
  for (; i < n; i += stride) {
    v4f f0 = *(const v4f*)(src + i);
    v4f f1 = *(const v4f*)(src + i + 4);
    v8bf o;
    o[0] = (bf16)f0[0]; o[1] = (bf16)f0[1]; o[2] = (bf16)f0[2]; o[3] = (bf16)f0[3];
    o[4] = (bf16)f1[0]; o[5] = (bf16)f1[1]; o[6] = (bf16)f1[2]; o[7] = (bf16)f1[3];
    *(v8bf*)(dst + i) = o;
  }
}

// ---------------------------------------------------------------------------
// 1) Routing: softmax -> top-2 -> renormalize
// ---------------------------------------------------------------------------
__global__ void route_kernel(const float* __restrict__ logits,
                             int* __restrict__ tki, float* __restrict__ tkw) {
  int t = blockIdx.x * blockDim.x + threadIdx.x;
  if (t >= TOK) return;
  float l[NEXP];
#pragma unroll
  for (int e = 0; e < NEXP; ++e) l[e] = logits[t * NEXP + e];
  float m = l[0];
#pragma unroll
  for (int e = 1; e < NEXP; ++e) m = fmaxf(m, l[e]);
  float p[NEXP];
#pragma unroll
  for (int e = 0; e < NEXP; ++e) p[e] = __expf(l[e] - m);
  int i0 = 0; float v0 = p[0];
#pragma unroll
  for (int e = 1; e < NEXP; ++e) if (p[e] > v0) { v0 = p[e]; i0 = e; }
  int i1 = -1; float v1 = -1.0f;
#pragma unroll
  for (int e = 0; e < NEXP; ++e)
    if (e != i0 && p[e] > v1) { v1 = p[e]; i1 = e; }
  const float inv = 1.0f / (v0 + v1);  // softmax denom cancels in the ratio
  tki[2 * t] = i0;      tki[2 * t + 1] = i1;
  tkw[2 * t] = v0 * inv; tkw[2 * t + 1] = v1 * inv;
}

// ---------------------------------------------------------------------------
// 2) Deterministic counting sort: per-expert token lists + token->slot map
// ---------------------------------------------------------------------------
__global__ void build_kernel(const int* __restrict__ tki,
                             int* __restrict__ counts, int* __restrict__ offs,
                             int* __restrict__ perm, int* __restrict__ slotmap) {
  __shared__ int scnt[NEXP], soff[NEXP];
  const int e = threadIdx.x;
  if (e < NEXP) {
    int c = 0;
    for (int t = 0; t < TOK; ++t) {
      if (tki[2 * t] == e) c++;
      if (tki[2 * t + 1] == e) c++;
    }
    scnt[e] = c;
  }
  __syncthreads();
  if (e == 0) {
    int o = 0;
    for (int i = 0; i < NEXP; ++i) { soff[i] = o; o += scnt[i]; }
  }
  __syncthreads();
  if (e < NEXP) {
    counts[e] = scnt[e];
    offs[e] = soff[e];
    int pos = soff[e];
    for (int t = 0; t < TOK; ++t) {
#pragma unroll
      for (int s = 0; s < 2; ++s) {
        if (tki[2 * t + s] == e) {
          perm[pos] = t;
          slotmap[2 * t + s] = pos;
          pos++;
        }
      }
    }
  }
}

// ---------------------------------------------------------------------------
// 3) GEMM1 + SwiGLU: h[slot, n] = silu(x·W1gT) * (x·W1uT)   (bf16 out)
//    Block tile 128(M) x 64(N of I); gate & up columns paired in one block.
//    256 threads = 8 wave32 (wave grid 4x2, 32x32 tiles). Ping/pong LDS fed
//    by global_load_async_to_lds_b128 with STATIC buffer addresses (K loop
//    unrolled x2) -> no per-iteration buffer-select cndmask chains.
// ---------------------------------------------------------------------------
__global__ __launch_bounds__(256)
void gemm1_kernel(const bf16* __restrict__ xb, const bf16* __restrict__ w1b,
                  const int* __restrict__ perm, const int* __restrict__ counts,
                  const int* __restrict__ offs, bf16* __restrict__ hbuf) {
  const int e = blockIdx.z;
  const int cnt = counts[e];
  const int m0 = blockIdx.y * 128;
  if (m0 >= cnt) return;                     // block-uniform: EXEC stays full
  const int n0 = blockIdx.x * 64;
  const int base = offs[e];
  const bf16* w1e = w1b + (size_t)e * (2 * IMID) * HID;
  const int* perm_e = perm + base;

  __shared__ __attribute__((aligned(16))) bf16 As[2][128 * LDT];
  __shared__ __attribute__((aligned(16))) bf16 Bg[2][64 * LDT];
  __shared__ __attribute__((aligned(16))) bf16 Bu[2][64 * LDT];

  const int tid = threadIdx.x;
  const int lane = tid & 31;
  const int wave = tid >> 5;
  const int wr = (wave >> 1) * 32;           // 0,32,64,96
  const int wc = (wave & 1) * 32;            // 0,32

  // A staging: 128x32 = 512 16B-chunks -> 2 per thread. Rows past cnt are
  // clamped to a valid slot: the garbage only lands in C rows that the
  // epilogue masks out, so the load path stays branch-free.
  const int c0 = tid,        r0 = c0 >> 2, kc0 = (c0 & 3) << 3;
  const int c1 = tid + 256,  r1 = c1 >> 2, kc1 = (c1 & 3) << 3;
  int s0i = m0 + r0; if (s0i >= cnt) s0i = cnt - 1;
  int s1i = m0 + r1; if (s1i >= cnt) s1i = cnt - 1;
  const bf16* pA0 = xb + (size_t)perm_e[s0i] * HID + kc0;
  const bf16* pA1 = xb + (size_t)perm_e[s1i] * HID + kc1;
  const unsigned lA0_0 = lds_off(&As[0][r0 * LDT + kc0]);
  const unsigned lA0_1 = lds_off(&As[0][r1 * LDT + kc1]);
  const unsigned lA1_0 = lds_off(&As[1][r0 * LDT + kc0]);
  const unsigned lA1_1 = lds_off(&As[1][r1 * LDT + kc1]);

  // B staging: 64x32 = 256 chunks -> 1 per thread per tile
  const int rB = tid >> 2;
  const int kcB = (tid & 3) << 3;
  const bf16* pG = w1e + (size_t)(n0 + rB) * HID + kcB;
  const bf16* pU = w1e + (size_t)(IMID + n0 + rB) * HID + kcB;
  const unsigned lG0 = lds_off(&Bg[0][rB * LDT + kcB]);
  const unsigned lG1 = lds_off(&Bg[1][rB * LDT + kcB]);
  const unsigned lU0 = lds_off(&Bu[0][rB * LDT + kcB]);
  const unsigned lU1 = lds_off(&Bu[1][rB * LDT + kcB]);

  v8f ag[2][2] = {};
  v8f au[2][2] = {};

  auto compute = [&](const bf16* At, const bf16* Bgt, const bf16* But) {
    v16bf a0 = frag_a(At, wr + 0, lane);
    v16bf a1 = frag_a(At, wr + 16, lane);
    v16bf g0 = frag_b(Bgt, wc + 0, lane);
    v16bf g1 = frag_b(Bgt, wc + 16, lane);
    v16bf u0 = frag_b(But, wc + 0, lane);
    v16bf u1 = frag_b(But, wc + 16, lane);
    ag[0][0] = mma_bf16(a0, g0, ag[0][0]);
    ag[0][1] = mma_bf16(a0, g1, ag[0][1]);
    ag[1][0] = mma_bf16(a1, g0, ag[1][0]);
    ag[1][1] = mma_bf16(a1, g1, ag[1][1]);
    au[0][0] = mma_bf16(a0, u0, au[0][0]);
    au[0][1] = mma_bf16(a0, u1, au[0][1]);
    au[1][0] = mma_bf16(a1, u0, au[1][0]);
    au[1][1] = mma_bf16(a1, u1, au[1][1]);
  };

  // Prologue: fill buffer 0 for k0 = 0
  async_ld16(lA0_0, pA0);
  async_ld16(lA0_1, pA1);
  async_ld16(lG0, pG);
  async_ld16(lU0, pU);

#pragma unroll 1
  for (int k0 = 0; k0 < HID; k0 += 64) {
    // ---- phase 0: compute buffer 0 (K chunk k0), prefetch into buffer 1 ----
    wait_async0();        // my writes into buf0 done
    __syncthreads();      // all writes done; all reads of buf1 done
    {
      const int k1 = k0 + 32;        // always < HID (HID/32 is even)
      async_ld16(lA1_0, pA0 + k1);
      async_ld16(lA1_1, pA1 + k1);
      async_ld16(lG1, pG + k1);
      async_ld16(lU1, pU + k1);
    }
    compute(As[0], Bg[0], Bu[0]);

    // ---- phase 1: compute buffer 1 (K chunk k0+32), prefetch into buffer 0 --
    wait_async0();
    __syncthreads();
    if (k0 + 64 < HID) {
      const int k2 = k0 + 64;
      async_ld16(lA0_0, pA0 + k2);
      async_ld16(lA0_1, pA1 + k2);
      async_ld16(lG0, pG + k2);
      async_ld16(lU0, pU + k2);
    }
    compute(As[1], Bg[1], Bu[1]);
  }

  // Epilogue: C layout -> lane n = lane&15, row = vgpr + 8*(lane>=16)
  const int mhi = (lane >> 4) << 3;
  const int nlo = lane & 15;
#pragma unroll
  for (int i = 0; i < 2; ++i)
#pragma unroll
    for (int j = 0; j < 2; ++j)
#pragma unroll
      for (int v = 0; v < 8; ++v) {
        const int row = m0 + wr + i * 16 + mhi + v;
        if (row < cnt) {
          const int col = n0 + wc + j * 16 + nlo;
          const float g = ag[i][j][v];
          const float u = au[i][j][v];
          const float s = g / (1.0f + __expf(-g));   // silu
          hbuf[(size_t)(base + row) * IMID + col] = (bf16)(s * u);
        }
      }
}

// ---------------------------------------------------------------------------
// 4) GEMM2: y[slot, H] = h · W2^T   (per-slot fp32 out, weights applied later)
// ---------------------------------------------------------------------------
__global__ __launch_bounds__(256)
void gemm2_kernel(const bf16* __restrict__ hbuf, const bf16* __restrict__ w2b,
                  const int* __restrict__ counts, const int* __restrict__ offs,
                  float* __restrict__ ybuf) {
  const int e = blockIdx.z;
  const int cnt = counts[e];
  const int m0 = blockIdx.y * 128;
  if (m0 >= cnt) return;
  const int n0 = blockIdx.x * 64;
  const int base = offs[e];
  const bf16* w2e = w2b + (size_t)e * HID * IMID;

  __shared__ __attribute__((aligned(16))) bf16 As[2][128 * LDT];
  __shared__ __attribute__((aligned(16))) bf16 Bs[2][64 * LDT];

  const int tid = threadIdx.x;
  const int lane = tid & 31;
  const int wave = tid >> 5;
  const int wr = (wave >> 1) * 32;
  const int wc = (wave & 1) * 32;

  const int c0 = tid,        r0 = c0 >> 2, kc0 = (c0 & 3) << 3;
  const int c1 = tid + 256,  r1 = c1 >> 2, kc1 = (c1 & 3) << 3;
  int s0i = m0 + r0; if (s0i >= cnt) s0i = cnt - 1;   // clamp; masked later
  int s1i = m0 + r1; if (s1i >= cnt) s1i = cnt - 1;
  const bf16* pA0 = hbuf + (size_t)(base + s0i) * IMID + kc0;
  const bf16* pA1 = hbuf + (size_t)(base + s1i) * IMID + kc1;
  const unsigned lA0_0 = lds_off(&As[0][r0 * LDT + kc0]);
  const unsigned lA0_1 = lds_off(&As[0][r1 * LDT + kc1]);
  const unsigned lA1_0 = lds_off(&As[1][r0 * LDT + kc0]);
  const unsigned lA1_1 = lds_off(&As[1][r1 * LDT + kc1]);

  const int rB = tid >> 2;
  const int kcB = (tid & 3) << 3;
  const bf16* pB = w2e + (size_t)(n0 + rB) * IMID + kcB;
  const unsigned lB0 = lds_off(&Bs[0][rB * LDT + kcB]);
  const unsigned lB1 = lds_off(&Bs[1][rB * LDT + kcB]);

  v8f acc[2][2] = {};

  auto compute = [&](const bf16* At, const bf16* Bt) {
    v16bf a0 = frag_a(At, wr + 0, lane);
    v16bf a1 = frag_a(At, wr + 16, lane);
    v16bf b0 = frag_b(Bt, wc + 0, lane);
    v16bf b1 = frag_b(Bt, wc + 16, lane);
    acc[0][0] = mma_bf16(a0, b0, acc[0][0]);
    acc[0][1] = mma_bf16(a0, b1, acc[0][1]);
    acc[1][0] = mma_bf16(a1, b0, acc[1][0]);
    acc[1][1] = mma_bf16(a1, b1, acc[1][1]);
  };

  async_ld16(lA0_0, pA0);
  async_ld16(lA0_1, pA1);
  async_ld16(lB0, pB);

#pragma unroll 1
  for (int k0 = 0; k0 < IMID; k0 += 64) {
    wait_async0();
    __syncthreads();
    {
      const int k1 = k0 + 32;        // always < IMID (IMID/32 is even)
      async_ld16(lA1_0, pA0 + k1);
      async_ld16(lA1_1, pA1 + k1);
      async_ld16(lB1, pB + k1);
    }
    compute(As[0], Bs[0]);

    wait_async0();
    __syncthreads();
    if (k0 + 64 < IMID) {
      const int k2 = k0 + 64;
      async_ld16(lA0_0, pA0 + k2);
      async_ld16(lA0_1, pA1 + k2);
      async_ld16(lB0, pB + k2);
    }
    compute(As[1], Bs[1]);
  }

  const int mhi = (lane >> 4) << 3;
  const int nlo = lane & 15;
#pragma unroll
  for (int i = 0; i < 2; ++i)
#pragma unroll
    for (int j = 0; j < 2; ++j)
#pragma unroll
      for (int v = 0; v < 8; ++v) {
        const int row = m0 + wr + i * 16 + mhi + v;
        if (row < cnt) {
          const int col = n0 + wc + j * 16 + nlo;
          ybuf[(size_t)(base + row) * HID + col] = acc[i][j][v];
        }
      }
}

// ---------------------------------------------------------------------------
// 5) Combine: out[t] = w0 * y[slot0] + w1 * y[slot1]  (deterministic, no atomics)
// ---------------------------------------------------------------------------
__global__ void combine_kernel(const float* __restrict__ ybuf,
                               const float* __restrict__ tkw,
                               const int* __restrict__ slotmap,
                               float* __restrict__ out) {
  const size_t n = (size_t)TOK * HID;
  size_t i = (size_t)blockIdx.x * blockDim.x + threadIdx.x;
  const size_t stride = (size_t)gridDim.x * blockDim.x;
  for (; i < n; i += stride) {
    const int t = (int)(i / HID);
    const int c = (int)(i % HID);
    const int s0 = slotmap[2 * t];
    const int s1 = slotmap[2 * t + 1];
    out[i] = tkw[2 * t] * ybuf[(size_t)s0 * HID + c] +
             tkw[2 * t + 1] * ybuf[(size_t)s1 * HID + c];
  }
}

// ---------------------------------------------------------------------------
// Host launcher
// ---------------------------------------------------------------------------
extern "C" void kernel_launch(void* const* d_in, const int* in_sizes, int n_in,
                              void* d_out, int out_size, void* d_ws, size_t ws_size,
                              hipStream_t stream) {
  (void)in_sizes; (void)n_in; (void)out_size; (void)ws_size;
  const float* x      = (const float*)d_in[0];   // [T, H]
  const float* logits = (const float*)d_in[1];   // [T, E]
  const float* w1     = (const float*)d_in[2];   // [E, 2I, H]
  const float* w2     = (const float*)d_in[3];   // [E, H, I]
  float* out = (float*)d_out;                    // [T, H]

  const size_t xn  = (size_t)TOK * HID;
  const size_t w1n = (size_t)NEXP * 2 * IMID * HID;
  const size_t w2n = (size_t)NEXP * HID * IMID;

  char* p = (char*)d_ws;
  auto take = [&p](size_t bytes) {
    char* q = p;
    p += (bytes + 255) & ~(size_t)255;
    return q;
  };
  int*   tki     = (int*)  take((size_t)2 * TOK * sizeof(int));
  float* tkw     = (float*)take((size_t)2 * TOK * sizeof(float));
  int*   counts  = (int*)  take(NEXP * sizeof(int));
  int*   offs    = (int*)  take(NEXP * sizeof(int));
  int*   perm    = (int*)  take((size_t)2 * TOK * sizeof(int));
  int*   slotmap = (int*)  take((size_t)2 * TOK * sizeof(int));
  bf16*  xb      = (bf16*) take(xn * sizeof(bf16));                       // ~17 MB
  bf16*  w1b     = (bf16*) take(w1n * sizeof(bf16));                      // ~369 MB
  bf16*  w2b     = (bf16*) take(w2n * sizeof(bf16));                      // ~185 MB
  bf16*  hbuf    = (bf16*) take((size_t)2 * TOK * IMID * sizeof(bf16));   // ~92 MB
  float* ybuf    = (float*)take((size_t)2 * TOK * HID * sizeof(float));   // ~67 MB

  route_kernel<<<TOK / 256, 256, 0, stream>>>(logits, tki, tkw);
  build_kernel<<<1, 32, 0, stream>>>(tki, counts, offs, perm, slotmap);

  // One-time bf16 conversions (bandwidth-bound, ~1.7 GB total traffic)
  cvt_kernel<<<2048, 256, 0, stream>>>(x, xb, xn);
  cvt_kernel<<<4096, 256, 0, stream>>>(w1, w1b, w1n);
  cvt_kernel<<<4096, 256, 0, stream>>>(w2, w2b, w2n);

  dim3 g1(IMID / 64, TOK / 128, NEXP);   // 88 x 32 x 8, early-exit past count
  gemm1_kernel<<<g1, 256, 0, stream>>>(xb, w1b, perm, counts, offs, hbuf);

  dim3 g2(HID / 64, TOK / 128, NEXP);    // 32 x 32 x 8
  gemm2_kernel<<<g2, 256, 0, stream>>>(hbuf, w2b, counts, offs, ybuf);

  combine_kernel<<<2048, 256, 0, stream>>>(ybuf, tkw, slotmap, out);
}
